// InstanceDiscriminationLoss_11879879544580
// MI455X (gfx1250) — matrix operand order repaired and checked by reading, not compiled, not run on hardware
//
#include <hip/hip_runtime.h>
#include <hip/hip_bf16.h>

// ---------------------------------------------------------------------------
// InstanceDiscriminationLoss for MI455X (gfx1250, wave32).
//
// Roofline: dominated by the 512 MB random gather of memory_bank rows for the
// negatives (HBM-bound, ~22us @ 23.3 TB/s). The only real GEMM
// (outputs @ W.T, 256x2048x128) is mapped onto v_wmma_f32_16x16x32_bf16 with
// a bf16 hi/lo split (3 WMMAs per K-chunk) for ~fp32 accuracy.
// ---------------------------------------------------------------------------

typedef __bf16 bf16_t;
typedef bf16_t bf16x8  __attribute__((ext_vector_type(8)));
typedef bf16_t bf16x16 __attribute__((ext_vector_type(16)));
typedef float  f32x8   __attribute__((ext_vector_type(8)));

#define N_BANK  1280000
#define D_DIM   128
#define DIN     2048
#define B_BATCH 256
#define M_NEG   4096

static constexpr float INV_TAU = 1.0f / 0.07f;
static constexpr float GAMMA_C = 0.5f;

// ----------------------------- wave reductions -----------------------------
__device__ __forceinline__ float wave_reduce_sum(float v) {
  v += __shfl_xor(v, 16, 32);
  v += __shfl_xor(v, 8, 32);
  v += __shfl_xor(v, 4, 32);
  v += __shfl_xor(v, 2, 32);
  v += __shfl_xor(v, 1, 32);
  return v;
}
__device__ __forceinline__ float wave_reduce_max(float v) {
  v = fmaxf(v, __shfl_xor(v, 16, 32));
  v = fmaxf(v, __shfl_xor(v, 8, 32));
  v = fmaxf(v, __shfl_xor(v, 4, 32));
  v = fmaxf(v, __shfl_xor(v, 2, 32));
  v = fmaxf(v, __shfl_xor(v, 1, 32));
  return v;
}

// ------------------- kernel 1: f32 -> bf16 hi/lo split ---------------------
__global__ __launch_bounds__(256) void split_bf16_kernel(
    const float* __restrict__ x, bf16_t* __restrict__ hi,
    bf16_t* __restrict__ lo, int n) {
  int i = blockIdx.x * blockDim.x + threadIdx.x;
  if (i < n) {
    float v = x[i];
    bf16_t h = (bf16_t)v;
    hi[i] = h;
    lo[i] = (bf16_t)(v - (float)h);
  }
}

// --------------- kernel 2: WMMA GEMM  C = A @ W^T + bias -------------------
// A = outputs (256 x 2048), W = (128 x 2048) row-major; both are K-major rows,
// which matches both the WMMA A layout (rows) and B layout (columns).
// One wave per 16x16 output tile; 128 tiles total.
__global__ __launch_bounds__(32) void gemm_bf16x3_wmma_kernel(
    const bf16_t* __restrict__ Ahi, const bf16_t* __restrict__ Alo,
    const bf16_t* __restrict__ Whi, const bf16_t* __restrict__ Wlo,
    const float* __restrict__ bias, float* __restrict__ out) {
  const int tile = blockIdx.x;       // 0..127
  const int tm   = tile >> 3;        // 16 row tiles (B=256)
  const int tn   = tile & 7;         // 8 col tiles (D=128)
  const int lane = threadIdx.x & 31;
  const int r    = lane & 15;
  const int sel  = lane >> 4;        // lane group 0/1

  const bf16_t* arh = Ahi + (size_t)(tm * 16 + r) * DIN;
  const bf16_t* arl = Alo + (size_t)(tm * 16 + r) * DIN;
  const bf16_t* brh = Whi + (size_t)(tn * 16 + r) * DIN;
  const bf16_t* brl = Wlo + (size_t)(tn * 16 + r) * DIN;

  f32x8 acc = {0.f, 0.f, 0.f, 0.f, 0.f, 0.f, 0.f, 0.f};

  for (int k0 = 0; k0 < DIN; k0 += 32) {
    // A layout: lanes 0-15 hold K {0..7,16..23}; lanes 16-31 hold {8..15,24..31}
    const int oa0 = k0 + sel * 8;
    const int oa1 = k0 + 16 + sel * 8;
    bf16x8 ah0 = *(const bf16x8*)(arh + oa0);
    bf16x8 ah1 = *(const bf16x8*)(arh + oa1);
    bf16x8 al0 = *(const bf16x8*)(arl + oa0);
    bf16x8 al1 = *(const bf16x8*)(arl + oa1);
    bf16x16 a_hi = __builtin_shufflevector(ah0, ah1, 0, 1, 2, 3, 4, 5, 6, 7,
                                           8, 9, 10, 11, 12, 13, 14, 15);
    bf16x16 a_lo = __builtin_shufflevector(al0, al1, 0, 1, 2, 3, 4, 5, 6, 7,
                                           8, 9, 10, 11, 12, 13, 14, 15);
    // B layout: lane holds column r, same K order as A
    bf16x8 bh0 = *(const bf16x8*)(brh + oa0);
    bf16x8 bh1 = *(const bf16x8*)(brh + oa1);
    bf16x8 bl0 = *(const bf16x8*)(brl + oa0);
    bf16x8 bl1 = *(const bf16x8*)(brl + oa1);
    bf16x16 b_hi = __builtin_shufflevector(bh0, bh1, 0, 1, 2, 3, 4, 5, 6, 7,
                                           8, 9, 10, 11, 12, 13, 14, 15);
    bf16x16 b_lo = __builtin_shufflevector(bl0, bl1, 0, 1, 2, 3, 4, 5, 6, 7,
                                           8, 9, 10, 11, 12, 13, 14, 15);

    // bf16x3: hi*hi + hi*lo + lo*hi (f32 accumulate) ~ fp32 GEMM
    acc = __builtin_amdgcn_wmma_f32_16x16x32_bf16(false, a_hi, false, b_hi,
                                                  (short)0, acc, false, false);
    acc = __builtin_amdgcn_wmma_f32_16x16x32_bf16(false, a_hi, false, b_lo,
                                                  (short)0, acc, false, false);
    acc = __builtin_amdgcn_wmma_f32_16x16x32_bf16(false, a_lo, false, b_hi,
                                                  (short)0, acc, false, false);
  }

  // C/D layout: lane L, VGPR v -> row M = v + 8*(L/16), col N = L%16
  const int   col   = tn * 16 + r;
  const float bj    = bias[col];
  const int   rbase = tm * 16 + sel * 8;
#pragma unroll
  for (int v = 0; v < 8; ++v)
    out[(size_t)(rbase + v) * D_DIM + col] = acc[v] + bj;
}

// ------------------- kernel 3: row-wise L2 normalize -----------------------
// one wave per row of emb_raw (256 rows x 128)
__global__ __launch_bounds__(256) void normalize_rows_kernel(
    const float* __restrict__ raw, float* __restrict__ emb) {
  const int lane = threadIdx.x & 31;
  const int wave = threadIdx.x >> 5;
  const int row  = blockIdx.x * 8 + wave;
  const float4 v = *(const float4*)(raw + (size_t)row * D_DIM + lane * 4);
  float s = v.x * v.x + v.y * v.y + v.z * v.z + v.w * v.w;
  s = wave_reduce_sum(s);
  const float scale = 1.0f / fmaxf(sqrtf(s), 1e-12f);
  float4 o;
  o.x = v.x * scale; o.y = v.y * scale; o.z = v.z * scale; o.w = v.w * scale;
  *(float4*)(emb + (size_t)row * D_DIM + lane * 4) = o;
}

// ---------------- kernel 4: pos logits + EMA entries -----------------------
// one wave per batch element
__global__ __launch_bounds__(256) void pos_entries_kernel(
    const float* __restrict__ emb, const float* __restrict__ mb,
    const int* __restrict__ indices, float* __restrict__ pos,
    float* __restrict__ entries_out) {
  const int lane = threadIdx.x & 31;
  const int wave = threadIdx.x >> 5;
  const int b    = blockIdx.x * 8 + wave;
  const int idx  = indices[b];
  const float4 e = *(const float4*)(emb + (size_t)b * D_DIM + lane * 4);
  const float4 p = *(const float4*)(mb + (size_t)idx * D_DIM + lane * 4);

  float d = e.x * p.x + e.y * p.y + e.z * p.z + e.w * p.w;
  d = wave_reduce_sum(d);
  if (lane == 0) pos[b] = d * INV_TAU;

  float4 ent;
  ent.x = GAMMA_C * p.x + (1.0f - GAMMA_C) * e.x;
  ent.y = GAMMA_C * p.y + (1.0f - GAMMA_C) * e.y;
  ent.z = GAMMA_C * p.z + (1.0f - GAMMA_C) * e.z;
  ent.w = GAMMA_C * p.w + (1.0f - GAMMA_C) * e.w;
  float s = ent.x * ent.x + ent.y * ent.y + ent.z * ent.z + ent.w * ent.w;
  s = wave_reduce_sum(s);
  const float scale = 1.0f / fmaxf(sqrtf(s), 1e-12f);
  ent.x *= scale; ent.y *= scale; ent.z *= scale; ent.w *= scale;
  *(float4*)(entries_out + (size_t)b * D_DIM + lane * 4) = ent;
}

// ------------------- kernel 5: negative logits (the big gather) ------------
// one block per batch element, 8 waves; each wave handles 512 negatives.
// A wave processes one memory_bank row at a time: float4/lane = fully
// coalesced 512B row read; dot via butterfly reduce. Indices fetched 32 at a
// time and broadcast with __shfl so the index stream is also coalesced.
__global__ __launch_bounds__(256) void neg_kernel(
    const float* __restrict__ emb, const float* __restrict__ mb,
    const int* __restrict__ ridx, float* __restrict__ negv) {
  const int b    = blockIdx.x;
  const int lane = threadIdx.x & 31;
  const int wave = threadIdx.x >> 5;
  const float4 e = *(const float4*)(emb + (size_t)b * D_DIM + lane * 4);
  const int base = b * M_NEG + wave * (M_NEG / 8);

  for (int m0 = 0; m0 < M_NEG / 8; m0 += 32) {
    const int idxv = ridx[base + m0 + lane];
    float keep = 0.0f;
#pragma unroll 4
    for (int i = 0; i < 32; ++i) {
      const int idx = __shfl(idxv, i, 32);
      const float4 v = *(const float4*)(mb + (size_t)idx * D_DIM + lane * 4);
      float p = e.x * v.x + e.y * v.y + e.z * v.z + e.w * v.w;
      p = wave_reduce_sum(p);            // all lanes get the dot
      if (lane == i) keep = p;
    }
    negv[base + m0 + lane] = keep * INV_TAU;
  }
}

// -------- kernel 6: per-b logsumexp + data/noise loss contributions --------
// one wave per batch element; neg row (16KB) re-read from cache, coalesced.
__global__ __launch_bounds__(256) void loss_terms_kernel(
    const float* __restrict__ negv, const float* __restrict__ pos,
    float* __restrict__ dterm, float* __restrict__ nterm) {
  const int lane = threadIdx.x & 31;
  const int wave = threadIdx.x >> 5;
  const int b    = blockIdx.x * 8 + wave;
  const float* row = negv + (size_t)b * M_NEG;

  float mx = -__builtin_inff();
  for (int j = lane; j < M_NEG; j += 32) mx = fmaxf(mx, row[j]);
  mx = wave_reduce_max(mx);

  float s = 0.0f;
  for (int j = lane; j < M_NEG; j += 32) s += expf(row[j] - mx);
  s = wave_reduce_sum(s);
  const float logC = mx + logf(s);       // logsumexp(neg[b,:])

  float acc = 0.0f;
  for (int j = lane; j < M_NEG; j += 32) {
    const float x  = row[j];
    const float m2 = fmaxf(x, logC);
    const float lad = m2 + log1pf(expf(-fabsf(x - logC)));  // logaddexp
    acc += logC - lad;                   // log_noise_prob
  }
  acc = wave_reduce_sum(acc);

  if (lane == 0) {
    const float pb   = pos[b];
    const float m3   = fmaxf(pb, logC);
    const float ladp = m3 + log1pf(expf(-fabsf(pb - logC)));
    dterm[b] = pb - ladp;                // log_data_prob
    nterm[b] = acc;
  }
}

// ------------------------- kernel 7: final reduce --------------------------
__global__ __launch_bounds__(256) void finalize_kernel(
    const float* __restrict__ dterm, const float* __restrict__ nterm,
    float* __restrict__ out) {
  __shared__ float sd[256];
  __shared__ float sn[256];
  const int t = threadIdx.x;
  sd[t] = dterm[t];
  sn[t] = nterm[t];
  __syncthreads();
  for (int s = 128; s > 0; s >>= 1) {
    if (t < s) { sd[t] += sd[t + s]; sn[t] += sn[t + s]; }
    __syncthreads();
  }
  if (t == 0) {
    const float dl = -sd[0] / (float)B_BATCH;
    const float nl = -sn[0] / (float)B_BATCH;
    out[0] = dl + nl;                        // loss
    out[1 + B_BATCH * D_DIM] = dl;           // data_loss
    out[2 + B_BATCH * D_DIM] = nl;           // noise_loss
  }
}

// ---------------------------------------------------------------------------
extern "C" void kernel_launch(void* const* d_in, const int* in_sizes, int n_in,
                              void* d_out, int out_size, void* d_ws,
                              size_t ws_size, hipStream_t stream) {
  (void)in_sizes; (void)n_in; (void)out_size; (void)ws_size;
  const float* outputs = (const float*)d_in[0];   // (256, 2048)
  const float* W       = (const float*)d_in[1];   // (128, 2048)
  const float* bias    = (const float*)d_in[2];   // (128,)
  const float* mb      = (const float*)d_in[3];   // (1280000, 128)
  const int*   indices = (const int*)d_in[4];     // (256,)
  const int*   ridx    = (const int*)d_in[5];     // (256, 4096)
  float*       out     = (float*)d_out;           // 1 + 32768 + 1 + 1 floats

  // workspace carve-out (all 1KB-aligned); total ~7.6 MB
  char*  ws  = (char*)d_ws;
  size_t off = 0;
  bf16_t* Ahi = (bf16_t*)(ws + off); off += (size_t)B_BATCH * DIN * 2;
  bf16_t* Alo = (bf16_t*)(ws + off); off += (size_t)B_BATCH * DIN * 2;
  bf16_t* Whi = (bf16_t*)(ws + off); off += (size_t)D_DIM * DIN * 2;
  bf16_t* Wlo = (bf16_t*)(ws + off); off += (size_t)D_DIM * DIN * 2;
  float* emb_raw = (float*)(ws + off); off += (size_t)B_BATCH * D_DIM * 4;
  float* emb     = (float*)(ws + off); off += (size_t)B_BATCH * D_DIM * 4;
  float* pos     = (float*)(ws + off); off += (size_t)B_BATCH * 4;
  float* dterm   = (float*)(ws + off); off += (size_t)B_BATCH * 4;
  float* nterm   = (float*)(ws + off); off += (size_t)B_BATCH * 4;
  float* negv    = (float*)(ws + off); off += (size_t)B_BATCH * M_NEG * 4;

  split_bf16_kernel<<<(B_BATCH * DIN + 255) / 256, 256, 0, stream>>>(
      outputs, Ahi, Alo, B_BATCH * DIN);
  split_bf16_kernel<<<(D_DIM * DIN + 255) / 256, 256, 0, stream>>>(
      W, Whi, Wlo, D_DIM * DIN);
  gemm_bf16x3_wmma_kernel<<<(B_BATCH / 16) * (D_DIM / 16), 32, 0, stream>>>(
      Ahi, Alo, Whi, Wlo, bias, emb_raw);
  normalize_rows_kernel<<<B_BATCH / 8, 256, 0, stream>>>(emb_raw, emb);
  pos_entries_kernel<<<B_BATCH / 8, 256, 0, stream>>>(emb, mb, indices, pos,
                                                      out + 1);
  neg_kernel<<<B_BATCH, 256, 0, stream>>>(emb, mb, ridx, negv);
  loss_terms_kernel<<<B_BATCH / 8, 256, 0, stream>>>(negv, pos, dterm, nterm);
  finalize_kernel<<<1, 256, 0, stream>>>(dterm, nterm, out);
}
